// GCNEncoder_5377299055294
// MI455X (gfx1250) — matrix operand order repaired and checked by reading, not compiled, not run on hardware
//
#include <hip/hip_runtime.h>

// ---------------------------------------------------------------------------
// GCN 2-layer encoder for MI455X (gfx1250, wave32, WMMA)
//   layer: out = scatter_add(norm_e * (xW)[src] -> dst) + dinv^2 * (xW) + b
// ---------------------------------------------------------------------------

typedef _Float16 v16h __attribute__((ext_vector_type(16)));
typedef _Float16 v8h  __attribute__((ext_vector_type(8)));
typedef float    v8f  __attribute__((ext_vector_type(8)));

#define NNODES 100000
#define NEDGES 1600000
#define CIN    128
#define CHID   128
#define COUT   64

// ---- degree / normalization ------------------------------------------------
__global__ void k_fill_ones(float* __restrict__ p, int n) {
  int i = blockIdx.x * blockDim.x + threadIdx.x;
  if (i < n) p[i] = 1.0f;
}

__global__ void k_deg_accum(const int* __restrict__ dst, float* __restrict__ deg, int e) {
  int i = blockIdx.x * blockDim.x + threadIdx.x;
  if (i < e) unsafeAtomicAdd(&deg[dst[i]], 1.0f);   // global_atomic_add_f32
}

__global__ void k_rsqrt_inplace(float* __restrict__ p, int n) {
  int i = blockIdx.x * blockDim.x + threadIdx.x;
  if (i < n) p[i] = rsqrtf(p[i]);
}

// ---- fp32 -> fp16 conversions ---------------------------------------------
__global__ void k_cvt_f16(const float* __restrict__ in, _Float16* __restrict__ out, int n) {
  int i = blockIdx.x * blockDim.x + threadIdx.x;
  if (i < n) out[i] = (_Float16)in[i];
}

__global__ void k_relu_cvt_f16(const float* __restrict__ in, _Float16* __restrict__ out, int n) {
  int i = blockIdx.x * blockDim.x + threadIdx.x;
  if (i < n) out[i] = (_Float16)fmaxf(in[i], 0.0f);
}

// ---- pack W (KxN, row-major fp32) into per-tile lane-major f16 B-fragments -
// B frag layout used here (contiguous k, same order as the A loads): lane l -> col n = l&15, K range
// 16*(l>>4)+0..15; half j of the v16h = K = 16*(l>>4)+j.
// Packed address: out[((kt*NT + nt)*32 + l)*16 + j]
__global__ void k_pack_w(const float* __restrict__ W, _Float16* __restrict__ out,
                         int N, int K) {
  int t = blockIdx.x * blockDim.x + threadIdx.x;
  if (t >= K * N) return;
  int NT   = N >> 4;
  int tile = t >> 9;          // 512 halfs per 32x16 tile
  int r    = t & 511;
  int l    = r >> 4;
  int j    = r & 15;
  int kt   = tile / NT;
  int nt   = tile - kt * NT;
  int k    = (kt << 5) + ((l >> 4) << 4) + j;
  int col  = (nt << 4) + (l & 15);
  out[t] = (_Float16)W[k * N + col];
}

// ---- WMMA GEMM + fused self-loop/bias epilogue -----------------------------
// H[m,n]   = (A @ W)[m,n]                      (fp32, feeds edge gather)
// Agg[m,n] = dinv[m]^2 * H[m,n] + bias[n]      (scatter-add target init)
// One wave = one 16x16 output tile. M % 16 == 0 so EXEC is all-1s at WMMA.
template <int N, int K>
__global__ void __launch_bounds__(256)
k_gcn_gemm_wmma(const _Float16* __restrict__ A, const _Float16* __restrict__ Bp,
                const float* __restrict__ dinv, const float* __restrict__ bias,
                float* __restrict__ H, float* __restrict__ Agg, int M) {
  constexpr int NT = N >> 4;
  constexpr int KT = K >> 5;
  const int wave = threadIdx.x >> 5;
  const int lane = threadIdx.x & 31;
  const int tile = blockIdx.x * 8 + wave;
  const int Mtiles = M >> 4;
  if (tile >= Mtiles * NT) return;               // uniform per wave
  const int mt = tile / NT;
  const int nt = tile - mt * NT;
  const int r0 = mt << 4;
  const int n0 = nt << 4;

  // A-fragment addressing: row = r0 + (lane&15); K slice selected by lane>>4
  const int m     = lane & 15;
  const int khalf = lane >> 4;
  const _Float16* arow = A + (size_t)(r0 + m) * K + khalf * 16;

  v8f c = {};
  union { v16h v; v8h h[2]; } a;
#pragma unroll
  for (int kt = 0; kt < KT; ++kt) {
    a.h[0] = *(const v8h*)(arow + kt * 32);       // K = kt*32 + khalf*16 + 0..7
    a.h[1] = *(const v8h*)(arow + kt * 32 + 8);   // K = kt*32 + khalf*16 + 8..15
    v16h b = *(const v16h*)(Bp + (((size_t)(kt * NT + nt)) * 32 + lane) * 16);
    c = __builtin_amdgcn_wmma_f32_16x16x32_f16(false, a.v, false, b,
                                               (short)0, c, false, false);
  }

  // C/D layout: lane n=lane&15; VGPR v -> row v + 8*(lane>>4)
  const int n  = lane & 15;
  const int mh = lane >> 4;
  const float bs = bias[n0 + n];
#pragma unroll
  for (int v = 0; v < 8; ++v) {
    const int row = r0 + v + mh * 8;
    const float val = c[v];
    const float dv  = dinv[row];
    const size_t idx = (size_t)row * N + n0 + n;
    H[idx]   = val;
    Agg[idx] = dv * dv * val + bs;
  }
}

// ---- edge scatter-add: agg[dst] += dinv[src]*dinv[dst] * h[src] ------------
// One thread = one edge x 4 channels (float4 gather, 4 fp32 atomics).
__global__ void k_edge_aggregate(const int* __restrict__ src, const int* __restrict__ dst,
                                 const float* __restrict__ dinv, const float* __restrict__ h,
                                 float* __restrict__ agg, int E, int cgShift) {
  const int tid = blockIdx.x * blockDim.x + threadIdx.x;
  if (tid >= (E << cgShift)) return;
  const int e  = tid >> cgShift;
  const int cg = tid & ((1 << cgShift) - 1);
  const int s = src[e];
  const int d = dst[e];
  const float w = dinv[s] * dinv[d];
  const float4 v = ((const float4*)h)[((size_t)s << cgShift) + cg];
  float* out = agg + ((((size_t)d << cgShift) + cg) << 2);
  unsafeAtomicAdd(out + 0, w * v.x);
  unsafeAtomicAdd(out + 1, w * v.y);
  unsafeAtomicAdd(out + 2, w * v.z);
  unsafeAtomicAdd(out + 3, w * v.w);
}

// ---------------------------------------------------------------------------
extern "C" void kernel_launch(void* const* d_in, const int* in_sizes, int n_in,
                              void* d_out, int out_size, void* d_ws, size_t ws_size,
                              hipStream_t stream) {
  const float* x  = (const float*)d_in[0];
  const int*   ei = (const int*)d_in[1];
  const float* W1 = (const float*)d_in[2];
  const float* b1 = (const float*)d_in[3];
  const float* W2 = (const float*)d_in[4];
  const float* b2 = (const float*)d_in[5];
  const int* src = ei;            // edge_index row 0
  const int* dst = ei + NEDGES;   // edge_index row 1
  float* out = (float*)d_out;     // [NNODES, COUT]

  // workspace carving (256B aligned); xh reused for relu-f16, h1 reused for h2
  char* ws = (char*)d_ws;
  size_t off = 0;
  auto carve = [&](size_t bytes) {
    void* p = ws + off;
    off += (bytes + 255) & ~(size_t)255;
    return p;
  };
  float*    dinv = (float*)   carve((size_t)NNODES * 4);
  _Float16* xh   = (_Float16*)carve((size_t)NNODES * CIN * 2);   // x_f16 / relu_f16
  _Float16* w1p  = (_Float16*)carve((size_t)CIN * CHID * 2);
  _Float16* w2p  = (_Float16*)carve((size_t)CHID * COUT * 2);
  float*    h1   = (float*)   carve((size_t)NNODES * CHID * 4);  // h1 / h2
  float*    agg1 = (float*)   carve((size_t)NNODES * CHID * 4);

  const int T = 256;
  auto blks = [](long n, int t) { return (unsigned)((n + t - 1) / t); };

  // 1) degrees -> dinv
  k_fill_ones   <<<blks(NNODES, T), T, 0, stream>>>(dinv, NNODES);
  k_deg_accum   <<<blks(NEDGES, T), T, 0, stream>>>(dst, dinv, NEDGES);
  k_rsqrt_inplace<<<blks(NNODES, T), T, 0, stream>>>(dinv, NNODES);

  // 2) operand prep
  k_cvt_f16<<<blks((long)NNODES * CIN, T), T, 0, stream>>>(x, xh, NNODES * CIN);
  k_pack_w <<<blks(CIN * CHID, T), T, 0, stream>>>(W1, w1p, CHID, CIN);
  k_pack_w <<<blks(CHID * COUT, T), T, 0, stream>>>(W2, w2p, COUT, CHID);

  // 3) layer 1: GEMM (WMMA) + fused self/bias init, edge scatter, ReLU->f16
  k_gcn_gemm_wmma<CHID, CIN><<<(NNODES / 16) * (CHID / 16) / 8, T, 0, stream>>>(
      xh, w1p, dinv, b1, h1, agg1, NNODES);
  k_edge_aggregate<<<blks((long)NEDGES * (CHID / 4), T), T, 0, stream>>>(
      src, dst, dinv, h1, agg1, NEDGES, 5);
  k_relu_cvt_f16<<<blks((long)NNODES * CHID, T), T, 0, stream>>>(
      agg1, xh, NNODES * CHID);

  // 4) layer 2: GEMM (WMMA) writes Agg init directly into d_out, edge scatter
  k_gcn_gemm_wmma<COUT, CHID><<<(NNODES / 16) * (COUT / 16) / 8, T, 0, stream>>>(
      xh, w2p, dinv, b2, h1, out, NNODES);
  k_edge_aggregate<<<blks((long)NEDGES * (COUT / 4), T), T, 0, stream>>>(
      src, dst, dinv, h1, out, NEDGES, 4);
}
